// MultiHeadGatedAxialAttentionWidth_63582695850407
// MI455X (gfx1250) — matrix operand order repaired and checked by reading, not compiled, not run on hardware
//
#include <hip/hip_runtime.h>
#include <hip/hip_bf16.h>
#include <math.h>

// MultiHeadGatedAxialAttentionWidth, MI455X / gfx1250 (wave32, WMMA).
// All GEMM work (qkv proj, qk, qrq, krk, P*V, P*rv, out proj) runs on
// v_wmma_f32_16x16x32_bf16. The position-dependent einsums are re-batched as
// dense GEMMs over the (n,i) axis per (head, j) so nothing falls back to VALU
// dot products. Workspace requirement: ~168.5 MiB.

#define NB    4
#define CCH   256
#define HH    64
#define WW    128
#define NHD   8
#define HDD   32
#define SS    (HH * WW)        /* 8192 spatial */
#define INV_D (1.0f / 16.0f)   /* 1/sqrt(C) */

typedef __bf16 bf16_t;
typedef __attribute__((ext_vector_type(16))) __bf16 v16bf;
typedef __attribute__((ext_vector_type(8)))  float  v8f;

#define V8F_ZERO {0.f, 0.f, 0.f, 0.f, 0.f, 0.f, 0.f, 0.f}

// Native conversions: clang lowers these to the hardware bf16 cvt path on
// gfx1250 (vs a 3-VALU integer RNE sequence).
__device__ __forceinline__ bf16_t f2b(float f) { return (bf16_t)f; }
__device__ __forceinline__ float  b2f(bf16_t b) { return (float)b; }

__device__ __forceinline__ int lane_id() { return (int)(threadIdx.x & 31u); }

// K index inside a 16x32 bf16 A operand for vector element e, lane-half khi.
// Lanes 0-15 hold K 0..15, lanes 16-31 hold K 16..31 (matches B fills).
__device__ __forceinline__ int a_kidx(int e, int khi) {
  return (khi << 4) | e;
}

__device__ __forceinline__ v8f wmma_bf16(v16bf a, v16bf b, v8f c) {
  return __builtin_amdgcn_wmma_f32_16x16x32_bf16(false, a, false, b,
                                                 (short)0, c, false, false);
}

// A tile (16x32): element (m,k) at p[m*ld + k]
__device__ __forceinline__ v16bf load_a(const bf16_t* p, int ld) {
  int l = lane_id(), m = l & 15, khi = l >> 4;
  v16bf a;
#pragma unroll
  for (int e = 0; e < 16; ++e) a[e] = p[m * ld + a_kidx(e, khi)];
  return a;
}
// B tile (32x16): element (k,n) at p[k*ldk + n*ldn]; lane = column N,
// lane-half selects K 0..15 vs 16..31.
__device__ __forceinline__ v16bf load_b(const bf16_t* p, int ldk, int ldn) {
  int l = lane_id(), n = l & 15, kb = (l >> 4) << 4;
  v16bf b;
#pragma unroll
  for (int e = 0; e < 16; ++e) b[e] = p[(kb + e) * ldk + n * ldn];
  return b;
}

// ---------------------------------------------------------------- prep:
// tanh + layout transposes of relative embeddings, tanh of gates, and one-time
// bf16 conversion of the four 256x256 weight matrices (so the GEMM inner loops
// load bf16 directly).
// rqt/rkt: [h][j][w][c]   (bias GEMM B operand, K=c contiguous)
// rvt    : [h][j][c][w]   (x2 GEMM B operand,  K=w contiguous)
__global__ void k_prep(const float* __restrict__ rq, const float* __restrict__ rk,
                       const float* __restrict__ rv,
                       const float* __restrict__ Gq, const float* __restrict__ Gk,
                       const float* __restrict__ Gv1, const float* __restrict__ Gv2,
                       const float* __restrict__ wq, const float* __restrict__ wk,
                       const float* __restrict__ wv, const float* __restrict__ wo,
                       bf16_t* __restrict__ rqt, bf16_t* __restrict__ rkt,
                       bf16_t* __restrict__ rvt, float* __restrict__ gates,
                       bf16_t* __restrict__ wqb, bf16_t* __restrict__ wkb,
                       bf16_t* __restrict__ wvb, bf16_t* __restrict__ wob) {
  const int total = NHD * HDD * WW * WW;  // 4194304, input layout [h][c][j][w]
  for (int t = blockIdx.x * blockDim.x + threadIdx.x; t < total;
       t += gridDim.x * blockDim.x) {
    int w = t & 127, j = (t >> 7) & 127, c = (t >> 14) & 31, h = t >> 19;
    int idx_jwc = ((h * WW + j) * WW + w) * HDD + c;
    rqt[idx_jwc] = f2b(tanhf(rq[t]));
    rkt[idx_jwc] = f2b(tanhf(rk[t]));
    rvt[((h * WW + j) * HDD + c) * WW + w] = f2b(tanhf(rv[t]));
  }
  const int wtotal = CCH * CCH;  // 65536 per weight
  for (int t = blockIdx.x * blockDim.x + threadIdx.x; t < wtotal;
       t += gridDim.x * blockDim.x) {
    wqb[t] = f2b(wq[t]);
    wkb[t] = f2b(wk[t]);
    wvb[t] = f2b(wv[t]);
    wob[t] = f2b(wo[t]);
  }
  if (blockIdx.x == 0 && threadIdx.x < 8) {
    int h = threadIdx.x;
    gates[h]      = tanhf(Gq[h]);
    gates[8 + h]  = tanhf(Gk[h]);
    gates[16 + h] = tanhf(Gv1[h]);
    gates[24 + h] = tanhf(Gv2[h]);
  }
}

// ---------------------------------------------------------------- qkv proj:
// q[n,o,s] = sum_c wq[o,c] x[n,c,s]; x tile staged once in LDS (bf16),
// reused for all three weights. Outputs scattered into attention layouts:
//   qb/kb: [h][width][n][i][c]  (c contiguous -> GEMM K dim)
//   vb   : [n][h][i][c][w]      (w contiguous -> x1 GEMM K dim)
__global__ void k_qkv(const float* __restrict__ x,
                      const bf16_t* __restrict__ wqb,
                      const bf16_t* __restrict__ wkb,
                      const bf16_t* __restrict__ wvb,
                      bf16_t* __restrict__ qb, bf16_t* __restrict__ kb,
                      bf16_t* __restrict__ vb) {
  __shared__ bf16_t xs[CCH][72];  // 64 spatial cols + pad, 36 KiB
  const int sb = blockIdx.x, n = blockIdx.y;
  const int s0 = sb * 64;
  const int tid = threadIdx.x;
  for (int rep = 0; rep < 64; ++rep) {
    int idx = rep * 256 + tid;
    int sl = idx & 63, c = idx >> 6;
    const float* xp = &x[((size_t)n * CCH + c) * SS + s0 + sl];
    if ((rep & 15) == 0) __builtin_prefetch(xp + 16 * SS, 0, 0);
    xs[c][sl] = f2b(*xp);
  }
  __syncthreads();
  const int wave = tid >> 5;
  const int nt = wave & 3;      // 16-col tile of the 64-col block
  const int mhalf = wave >> 2;  // 8 output-channel tiles each
  const int l = lane_id(), ncol = l & 15, mh8 = (l >> 4) << 3;
#pragma unroll
  for (int wsel = 0; wsel < 3; ++wsel) {
    const bf16_t* wmat = (wsel == 0) ? wqb : (wsel == 1) ? wkb : wvb;
    for (int mt = mhalf * 8; mt < mhalf * 8 + 8; ++mt) {
      v8f acc = V8F_ZERO;
      for (int kk = 0; kk < 8; ++kk) {
        v16bf a = load_a(wmat + mt * 16 * CCH + kk * 32, CCH);
        v16bf b = load_b(&xs[kk * 32][nt * 16], 72, 1);
        acc = wmma_bf16(a, b, acc);
      }
      const int s = s0 + nt * 16 + ncol;
      const int i = s >> 7, j = s & 127;
#pragma unroll
      for (int r = 0; r < 8; ++r) {
        int o = mt * 16 + r + mh8;
        int h = o >> 5, ch = o & 31;
        bf16_t val = f2b(acc[r]);
        if (wsel == 2) {
          vb[((((size_t)n * NHD + h) * HH + i) * HDD + ch) * WW + j] = val;
        } else {
          bf16_t* dst = (wsel == 0) ? qb : kb;
          dst[(((size_t)(h * WW + j) * NB + n) * HH + i) * HDD + ch] = val;
        }
      }
    }
  }
}

// ---------------------------------------------------------------- bias:
// per (h,j): bias[(n,i),w] = (gq * q[(n,i),:]@rqt[:,w] + gk * k@rkt) / D
// as two 256x32x128 GEMMs over the (n,i) batch. Output [n][h][i][j][w] bf16.
__global__ void k_bias(const bf16_t* __restrict__ qb, const bf16_t* __restrict__ kb,
                       const bf16_t* __restrict__ rqt, const bf16_t* __restrict__ rkt,
                       const float* __restrict__ gates, bf16_t* __restrict__ bias) {
  const int j = blockIdx.x, h = blockIdx.y;
  const float gq = gates[h] * INV_D, gk = gates[8 + h] * INV_D;
  const int wave = threadIdx.x >> 5;
  const int l = lane_id(), ncol = l & 15, mh8 = (l >> 4) << 3;
  const bf16_t* Aq = qb + (size_t)(h * WW + j) * (NB * HH * HDD);  // rows (n,i)
  const bf16_t* Ak = kb + (size_t)(h * WW + j) * (NB * HH * HDD);
  const bf16_t* Bq = rqt + (size_t)(h * WW + j) * (WW * HDD);      // [w][c]
  const bf16_t* Bk = rkt + (size_t)(h * WW + j) * (WW * HDD);
  for (int mt = wave * 2; mt < wave * 2 + 2; ++mt) {
    v16bf aq = load_a(Aq + mt * 16 * HDD, HDD);
    v16bf ak = load_a(Ak + mt * 16 * HDD, HDD);
    for (int nt = 0; nt < 8; ++nt) {
      v16bf bq = load_b(Bq + nt * 16 * HDD, 1, HDD);
      v16bf bk = load_b(Bk + nt * 16 * HDD, 1, HDD);
      v8f accq = V8F_ZERO, acck = V8F_ZERO;
      accq = wmma_bf16(aq, bq, accq);
      acck = wmma_bf16(ak, bk, acck);
      const int w = nt * 16 + ncol;
#pragma unroll
      for (int r = 0; r < 8; ++r) {
        int row = mt * 16 + r + mh8;
        int n = row >> 6, i = row & 63;
        bias[(((size_t)(n * NHD + h) * HH + i) * WW + j) * WW + w] =
            f2b(gq * accq[r] + gk * acck[r]);
      }
    }
  }
}

// ---------------------------------------------------------------- attention:
// per (n,h,i): S = q^T k / D + bias; row softmax in registers (shfl_xor over
// the 16 lanes sharing a row of the C/D layout); x1 = P @ V^T via WMMA with P
// staged per-wave in LDS. P also spilled to global (overwriting the bias
// buffer slice of this block - reads strictly precede writes) for the x2 pass.
__global__ void k_attn(const bf16_t* __restrict__ qb, const bf16_t* __restrict__ kb,
                       const bf16_t* __restrict__ vb,
                       const bf16_t* bias_in,   // aliases P_out: no restrict
                       const float* __restrict__ gates,
                       bf16_t* P_out, float* __restrict__ attout) {
  __shared__ bf16_t Pl[8][16][WW + 8];  // per-wave P tile, 34 KiB
  const int i = blockIdx.x, h = blockIdx.y, n = blockIdx.z;
  const int wave = threadIdx.x >> 5;  // = j-tile
  const int jt = wave;
  const int l = lane_id(), ncol = l & 15, khi = l >> 4, mh8 = khi << 3,
            m = l & 15;
  const float gv1 = gates[16 + h];
  const int RS = NB * HH * HDD;  // stride over width coord in qb/kb

  // A = q^T tile (16 j rows x 32 c)
  const bf16_t* qp = qb + (((size_t)(h * WW + jt * 16) * NB + n) * HH + i) * HDD;
  v16bf aq;
#pragma unroll
  for (int e = 0; e < 16; ++e) aq[e] = qp[(size_t)m * RS + a_kidx(e, khi)];

  v8f acc[8];
#pragma unroll
  for (int wt = 0; wt < 8; ++wt) {
    const bf16_t* kp =
        kb + (((size_t)(h * WW + wt * 16) * NB + n) * HH + i) * HDD;
    v16bf b;
#pragma unroll
    for (int e = 0; e < 16; ++e) b[e] = kp[(size_t)ncol * RS + (khi << 4) + e];
    v8f z = V8F_ZERO;
    acc[wt] = wmma_bf16(aq, b, z);
  }

  // scale + relative-position bias
  const bf16_t* bb = bias_in + ((size_t)(n * NHD + h) * HH + i) * (WW * WW);
#pragma unroll
  for (int wt = 0; wt < 8; ++wt)
#pragma unroll
    for (int r = 0; r < 8; ++r) {
      int jl = jt * 16 + r + mh8;
      int w = wt * 16 + ncol;
      acc[wt][r] = acc[wt][r] * INV_D + b2f(bb[(size_t)jl * WW + w]);
    }

  // register-resident softmax over w (8 tiles x 16 lanes per row)
  float rinv[8];
#pragma unroll
  for (int r = 0; r < 8; ++r) {
    float mx = -3.0e38f;
#pragma unroll
    for (int wt = 0; wt < 8; ++wt) mx = fmaxf(mx, acc[wt][r]);
    for (int msk = 1; msk < 16; msk <<= 1)
      mx = fmaxf(mx, __shfl_xor(mx, msk, 32));
    float sm = 0.f;
#pragma unroll
    for (int wt = 0; wt < 8; ++wt) {
      float ev = __expf(acc[wt][r] - mx);
      acc[wt][r] = ev;
      sm += ev;
    }
    for (int msk = 1; msk < 16; msk <<= 1) sm += __shfl_xor(sm, msk, 32);
    rinv[r] = 1.f / sm;
  }

  // normalize; stage P in LDS and spill to global for the x2 pass
  bf16_t* Pg = P_out + ((size_t)(n * NHD + h) * HH + i) * (WW * WW);
#pragma unroll
  for (int wt = 0; wt < 8; ++wt)
#pragma unroll
    for (int r = 0; r < 8; ++r) {
      int jl16 = r + mh8, w = wt * 16 + ncol;
      bf16_t pv = f2b(acc[wt][r] * rinv[r]);
      Pl[wave][jl16][w] = pv;
      Pg[(size_t)(jt * 16 + jl16) * WW + w] = pv;
    }
  __syncthreads();

  // x1 = P @ V^T : M=j, N=c(32 -> 2 tiles), K=w(128 -> 4 steps)
  const bf16_t* vp = vb + ((size_t)(n * NHD + h) * HH + i) * (HDD * WW);
  float* ob = attout + ((size_t)n * CCH + h * HDD) * SS + i * WW;
  for (int ct = 0; ct < 2; ++ct) {
    v8f accx = V8F_ZERO;
    for (int kk = 0; kk < 4; ++kk) {
      v16bf a = load_a(&Pl[wave][0][kk * 32], WW + 8);
      v16bf b;
#pragma unroll
      for (int e = 0; e < 16; ++e)
        b[e] = vp[(size_t)(ct * 16 + ncol) * WW + kk * 32 + (khi << 4) + e];
      accx = wmma_bf16(a, b, accx);
    }
#pragma unroll
    for (int r = 0; r < 8; ++r) {
      int jl = jt * 16 + r + mh8;
      int c = ct * 16 + ncol;
      ob[(size_t)c * SS + jl] = gv1 * accx[r];
    }
  }
}

// ---------------------------------------------------------------- x2:
// per (h,j): x2[(n,i),c] = sum_w P[(n,i),w] rvt[c,w] as a 256x128x32 GEMM;
// accumulate gv2*x2 into attout (disjoint elements per block).
__global__ void k_x2(const bf16_t* __restrict__ P, const bf16_t* __restrict__ rvt,
                     const float* __restrict__ gates, float* __restrict__ attout) {
  const int j = blockIdx.x, h = blockIdx.y;
  const float gv2 = gates[24 + h];
  const int wave = threadIdx.x >> 5;
  const int l = lane_id(), ncol = l & 15, khi = l >> 4, mh8 = khi << 3,
            m = l & 15;
  const bf16_t* Bb = rvt + (size_t)(h * WW + j) * (HDD * WW);  // [c][w]
  for (int mt = wave * 2; mt < wave * 2 + 2; ++mt) {
    const int row0 = mt * 16, n = row0 >> 6, i0 = row0 & 63;
    const bf16_t* Ab = P + (((size_t)(n * NHD + h) * HH + i0) * WW + j) * WW;
    v16bf a[4];
#pragma unroll
    for (int kk = 0; kk < 4; ++kk)
#pragma unroll
      for (int e = 0; e < 16; ++e)
        a[kk][e] = Ab[(size_t)m * (WW * WW) + kk * 32 + a_kidx(e, khi)];
    for (int ct = 0; ct < 2; ++ct) {
      v8f acc = V8F_ZERO;
#pragma unroll
      for (int kk = 0; kk < 4; ++kk) {
        v16bf b;
#pragma unroll
        for (int e = 0; e < 16; ++e)
          b[e] = Bb[(size_t)(ct * 16 + ncol) * WW + kk * 32 + (khi << 4) + e];
        acc = wmma_bf16(a[kk], b, acc);
      }
#pragma unroll
      for (int r = 0; r < 8; ++r) {
        int i = i0 + r + mh8;
        int c = ct * 16 + ncol;
        float* dst =
            attout + ((size_t)n * CCH + h * HDD + c) * SS + i * WW + j;
        *dst += gv2 * acc[r];
      }
    }
  }
}

// ---------------------------------------------------------------- out proj:
// out[n,o,s] = sum_c wout[o,c] attout[n,c,s]
__global__ void k_outproj(const bf16_t* __restrict__ wob,
                          const float* __restrict__ attout,
                          float* __restrict__ out) {
  __shared__ bf16_t xs[CCH][72];
  const int sb = blockIdx.x, n = blockIdx.y;
  const int s0 = sb * 64;
  const int tid = threadIdx.x;
  for (int rep = 0; rep < 64; ++rep) {
    int idx = rep * 256 + tid;
    int sl = idx & 63, c = idx >> 6;
    const float* ap = &attout[((size_t)n * CCH + c) * SS + s0 + sl];
    if ((rep & 15) == 0) __builtin_prefetch(ap + 16 * SS, 0, 0);
    xs[c][sl] = f2b(*ap);
  }
  __syncthreads();
  const int wave = tid >> 5;
  const int nt = wave & 3;
  const int mhalf = wave >> 2;
  const int l = lane_id(), ncol = l & 15, mh8 = (l >> 4) << 3;
  for (int mt = mhalf * 8; mt < mhalf * 8 + 8; ++mt) {
    v8f acc = V8F_ZERO;
    for (int kk = 0; kk < 8; ++kk) {
      v16bf a = load_a(wob + mt * 16 * CCH + kk * 32, CCH);
      v16bf b = load_b(&xs[kk * 32][nt * 16], 72, 1);
      acc = wmma_bf16(a, b, acc);
    }
    const int s = s0 + nt * 16 + ncol;
#pragma unroll
    for (int r = 0; r < 8; ++r) {
      int o = mt * 16 + r + mh8;
      out[((size_t)n * CCH + o) * SS + s] = acc[r];
    }
  }
}

extern "C" void kernel_launch(void* const* d_in, const int* in_sizes, int n_in,
                              void* d_out, int out_size, void* d_ws,
                              size_t ws_size, hipStream_t stream) {
  const float* x    = (const float*)d_in[0];
  const float* wq   = (const float*)d_in[1];
  const float* wk   = (const float*)d_in[2];
  const float* wv   = (const float*)d_in[3];
  const float* wout = (const float*)d_in[4];
  const float* rq   = (const float*)d_in[5];
  const float* rk   = (const float*)d_in[6];
  const float* rv   = (const float*)d_in[7];
  const float* Gq   = (const float*)d_in[8];
  const float* Gk   = (const float*)d_in[9];
  const float* Gv1  = (const float*)d_in[10];
  const float* Gv2  = (const float*)d_in[11];
  float* out = (float*)d_out;

  // Workspace map (bytes); total ~168.5 MiB.
  char* w = (char*)d_ws;
  const size_t SZ_QKV = (size_t)NB * CCH * SS * sizeof(bf16_t);        // 16 MiB
  const size_t SZ_R   = (size_t)NHD * HDD * WW * WW * sizeof(bf16_t);  //  8 MiB
  const size_t SZ_PB  = (size_t)NB * NHD * HH * WW * WW * sizeof(bf16_t); // 64 MiB
  const size_t SZ_AO  = (size_t)NB * CCH * SS * sizeof(float);         // 32 MiB
  const size_t SZ_W   = (size_t)CCH * CCH * sizeof(bf16_t);            // 128 KiB
  size_t off = 0;
  bf16_t* qb    = (bf16_t*)(w + off); off += SZ_QKV;
  bf16_t* kb    = (bf16_t*)(w + off); off += SZ_QKV;
  bf16_t* vb    = (bf16_t*)(w + off); off += SZ_QKV;
  bf16_t* rqt   = (bf16_t*)(w + off); off += SZ_R;
  bf16_t* rkt   = (bf16_t*)(w + off); off += SZ_R;
  bf16_t* rvt   = (bf16_t*)(w + off); off += SZ_R;
  bf16_t* biasP = (bf16_t*)(w + off); off += SZ_PB;  // bias, then reused as P
  float*  attout= (float*)(w + off);  off += SZ_AO;
  bf16_t* wqb   = (bf16_t*)(w + off); off += SZ_W;
  bf16_t* wkb   = (bf16_t*)(w + off); off += SZ_W;
  bf16_t* wvb   = (bf16_t*)(w + off); off += SZ_W;
  bf16_t* wob   = (bf16_t*)(w + off); off += SZ_W;
  float*  gates = (float*)(w + off);  off += 256;

  k_prep<<<dim3(2048), dim3(256), 0, stream>>>(rq, rk, rv, Gq, Gk, Gv1, Gv2,
                                               wq, wk, wv, wout,
                                               rqt, rkt, rvt, gates,
                                               wqb, wkb, wvb, wob);
  k_qkv<<<dim3(128, NB), dim3(256), 0, stream>>>(x, wqb, wkb, wvb, qb, kb, vb);
  k_bias<<<dim3(WW, NHD), dim3(256), 0, stream>>>(qb, kb, rqt, rkt, gates, biasP);
  k_attn<<<dim3(HH, NHD, NB), dim3(256), 0, stream>>>(qb, kb, vb, biasP, gates,
                                                      biasP, attout);
  k_x2<<<dim3(WW, NHD), dim3(256), 0, stream>>>(biasP, rvt, gates, attout);
  k_outproj<<<dim3(128, NB), dim3(256), 0, stream>>>(wob, attout, out);

  (void)in_sizes; (void)n_in; (void)out_size; (void)ws_size;
}
